// Attention_403726925957
// MI455X (gfx1250) — matrix operand order repaired and checked by reading, not compiled, not run on hardware
//
#include <hip/hip_runtime.h>

#define B_   2
#define S_   2048
#define DIM_ 4096
#define NH_  32
#define NKV_ 8
#define HD_  128

typedef __attribute__((ext_vector_type(16))) __bf16 v16bf;
typedef __attribute__((ext_vector_type(8)))  __bf16 v8bf;
typedef __attribute__((ext_vector_type(8)))  float  v8f;
typedef __attribute__((ext_vector_type(4)))  unsigned v4u;
typedef __attribute__((ext_vector_type(8)))  unsigned v8u;

__device__ __forceinline__ __bf16 f2bf(float f) {
  unsigned int u = __builtin_bit_cast(unsigned int, f);
  u += 0x7FFFu + ((u >> 16) & 1u);              // round-to-nearest-even
  unsigned short h = (unsigned short)(u >> 16);
  return __builtin_bit_cast(__bf16, h);
}

__device__ __forceinline__ v8f wmma_bf16(v16bf a, v16bf b, v8f c) {
  // (neg_a, A, neg_b, B, c_mod, C, reuse_a, reuse_b)
  return __builtin_amdgcn_wmma_f32_16x16x32_bf16(false, a, false, b, (short)0, c,
                                                 false, false);
}

// --------------------------------------------------------------- TDM load
// One Tensor-Data-Mover op moves a 128-row x 32-elem bf16 tile (64B rows)
// from global to LDS, padding each LDS row with 16B (4 DWORDs) so rows land
// at 80B stride (= our [128][40] __bf16 LDS layout). TENSORcnt-tracked.
__device__ __forceinline__ void tdm_load_tile(unsigned lds_addr,
                                              const __bf16* g,
                                              int stride_elems) {
  const unsigned long long ga = (unsigned long long)(uintptr_t)g;
  v4u g0;
  g0[0] = 1u;                                 // D# count=1 (valid), user mode
  g0[1] = lds_addr;                           // LDS destination (bytes)
  g0[2] = (unsigned)ga;                       // global_addr[31:0]
  g0[3] = (unsigned)(ga >> 32) | (2u << 30);  // global_addr[56:32] | type=2
  v8u g1;
  // data_size=2B, pad_enable, pad_interval=3 (16 DWORDs), pad_amount=3 (4 DW)
  g1[0] = (1u << 16) | (1u << 20) | (3u << 22) | (3u << 25);
  g1[1] = 32u << 16;                          // tensor_dim0 = 32 (lo16)
  g1[2] = 128u << 16;                         // tensor_dim1 = 128 (lo16)
  g1[3] = 32u << 16;                          // tile_dim0 = 32
  g1[4] = 128u;                               // tile_dim1 = 128, tile_dim2 = 0
  g1[5] = (unsigned)stride_elems;             // tensor_dim0_stride (lo32)
  g1[6] = 0u;                                 // stride hi16 / dim1_stride lo
  g1[7] = 0u;
  asm volatile("tensor_load_to_lds %0, %1" :: "s"(g0), "s"(g1) : "memory");
}

// ---------------------------------------------------------------- converts
__global__ void convert_bf16(const float* __restrict__ src,
                             __bf16* __restrict__ dst, long n) {
  long i = (long)blockIdx.x * blockDim.x + threadIdx.x;
  const long stride = (long)gridDim.x * blockDim.x;
  for (; i < n; i += stride) dst[i] = f2bf(src[i]);
}

// RoPE (interleaved pairs) + cast to bf16. src layout (b, s, h, 128).
__global__ void rope_to_bf16(const float* __restrict__ src,
                             const float* __restrict__ cs,
                             const float* __restrict__ sn,
                             __bf16* __restrict__ dst,
                             int nheads, long npairs) {
  long idx = (long)blockIdx.x * blockDim.x + threadIdx.x;
  if (idx >= npairs) return;
  const int d2 = (int)(idx & 63);
  long t = idx >> 6;
  const int hh = (int)(t % nheads); t /= nheads;
  const int s  = (int)(t % S_);
  const int b  = (int)(t / S_);
  const long base = ((long)(b * S_ + s) * nheads + hh) * HD_;
  const float tr = src[base + 2 * d2];
  const float ti = src[base + 2 * d2 + 1];
  const float c  = cs[s * 64 + d2];
  const float si = sn[s * 64 + d2];
  dst[base + 2 * d2]     = f2bf(tr * c - ti * si);
  dst[base + 2 * d2 + 1] = f2bf(tr * si + ti * c);
}

// V: cast + transpose to (b, kv_head, d, s) so PV B-fragments load contiguously.
__global__ void v_transpose_bf16(const float* __restrict__ src,
                                 __bf16* __restrict__ dst, long n) {
  long idx = (long)blockIdx.x * blockDim.x + threadIdx.x;
  if (idx >= n) return;
  const int d = (int)(idx & 127);
  long t = idx >> 7;
  const int hh = (int)(t & 7); t >>= 3;
  const int s  = (int)(t & (S_ - 1));
  const int b  = (int)(t >> 11);
  dst[((long)(b * NKV_ + hh) * HD_ + d) * S_ + s] = f2bf(src[idx]);
}

// ------------------------------------------------- GEMM: C = A(MxK) * W(NxK)^T
// 128x128 block tile, 8 waves, each wave 32x64 via 2x4 wmma_f32_16x16x32_bf16.
// Double-buffered LDS filled by the Tensor Data Mover (one op per tile,
// issued by wave 0, TENSORcnt-tracked, LDS rows padded by the TDM).
__global__ __launch_bounds__(256) void gemm_bf16_nt(
    const __bf16* __restrict__ A, const __bf16* __restrict__ W,
    float* __restrict__ C, int M, int N, int K) {
  __shared__ __align__(16) __bf16 As[2][128][40];
  __shared__ __align__(16) __bf16 Bs[2][128][40];
  const int tid  = threadIdx.x;
  const int lane = tid & 31;
  const int wid  = tid >> 5;
  const int wm   = wid & 3;       // wave row (32 rows each)
  const int wn   = wid >> 2;      // wave col (64 cols each)
  const int lrow = lane & 15;
  const int hi16 = lane >> 4;
  const long row0 = (long)blockIdx.y * 128;
  const long col0 = (long)blockIdx.x * 128;
  const v8f vzero = {0.f, 0.f, 0.f, 0.f, 0.f, 0.f, 0.f, 0.f};

  v8f acc[2][4];
#pragma unroll
  for (int i = 0; i < 2; ++i)
#pragma unroll
    for (int j = 0; j < 4; ++j) acc[i][j] = vzero;

  const __bf16* gA = A + row0 * (long)K;
  const __bf16* gB = W + col0 * (long)K;
  const unsigned ldsA[2] = {(unsigned)(uintptr_t)&As[0][0][0],
                            (unsigned)(uintptr_t)&As[1][0][0]};
  const unsigned ldsB[2] = {(unsigned)(uintptr_t)&Bs[0][0][0],
                            (unsigned)(uintptr_t)&Bs[1][0][0]};

  // Prologue: TDM-fill buffer 0 with K-tile 0 (wave 0 only; EXEC-independent).
  if (wid == 0) {
    tdm_load_tile(ldsA[0], gA, K);
    tdm_load_tile(ldsB[0], gB, K);
  }

  int cur = 0;
  for (int k0 = 0; k0 < K; k0 += 32) {
    if (wid == 0) __builtin_amdgcn_s_wait_tensorcnt(0);  // buffer `cur` landed
    __syncthreads();  // publish to all waves; prior reads already retired
    const int nxt = cur ^ 1;
    if (wid == 0 && k0 + 32 < K) {  // kick off next tile while we compute
      tdm_load_tile(ldsA[nxt], gA + k0 + 32, K);
      tdm_load_tile(ldsB[nxt], gB + k0 + 32, K);
    }

    v16bf af[2], bfv[4];
#pragma unroll
    for (int i = 0; i < 2; ++i) {
      const int r  = wm * 32 + i * 16 + lrow;
      const int kb = hi16 * 16;                      // K=0..7 / 16..23
      v8bf lo = *(const v8bf*)&As[cur][r][kb];
      v8bf hi = *(const v8bf*)&As[cur][r][kb + 8];   // K=8..15 / 24..31
      af[i] = __builtin_shufflevector(lo, hi, 0, 1, 2, 3, 4, 5, 6, 7,
                                      8, 9, 10, 11, 12, 13, 14, 15);
    }
#pragma unroll
    for (int j = 0; j < 4; ++j) {
      const int n = wn * 64 + j * 16 + lrow;
      bfv[j] = *(const v16bf*)&Bs[cur][n][hi16 * 16];  // contiguous K half
    }
#pragma unroll
    for (int i = 0; i < 2; ++i)
#pragma unroll
      for (int j = 0; j < 4; ++j)
        acc[i][j] = wmma_bf16(af[i], bfv[j], acc[i][j]);

    cur = nxt;
  }

#pragma unroll
  for (int i = 0; i < 2; ++i)
#pragma unroll
    for (int j = 0; j < 4; ++j) {
      const long col = col0 + wn * 64 + j * 16 + lrow;
#pragma unroll
      for (int r = 0; r < 8; ++r) {
        const long row = row0 + wm * 32 + i * 16 + r + hi16 * 8;
        C[row * (long)N + col] = acc[i][j][r];
      }
    }
  (void)M;
}

// ------------------------------------------ flash attention, one wave / 16 q
__global__ __launch_bounds__(256) void flash_attn_wmma(
    const __bf16* __restrict__ Q,   // (b, s, 32, 128) bf16, post-RoPE
    const __bf16* __restrict__ Kb,  // (b, s, 8, 128)  bf16, post-RoPE
    const __bf16* __restrict__ Vt,  // (b, 8, 128, s)  bf16 (transposed)
    __bf16* __restrict__ Ob) {      // (b, s, 4096)    bf16
  __shared__ __align__(16) __bf16 Ps[8][16][40];
  const int tid  = threadIdx.x;
  const int lane = tid & 31;
  const int wid  = tid >> 5;
  const int lrow = lane & 15;
  const int hi16 = lane >> 4;
  const int h  = blockIdx.y;
  const int b  = blockIdx.z;
  const int hk = h >> 2;                       // GQA: 4 Q heads per KV head
  const int qs = (blockIdx.x * 8 + wid) * 16;  // this wave's q-tile base

  // Preload Q fragments for the whole kernel: 4 head-dim chunks of 32.
  v16bf qf[4];
  const __bf16* qbase = Q + ((long)(b * S_ + qs) * NH_ + h) * HD_;
#pragma unroll
  for (int c = 0; c < 4; ++c) {
    const __bf16* p = qbase + (long)lrow * (NH_ * HD_) + c * 32 + hi16 * 16;
    v8bf lo = *(const v8bf*)p;
    v8bf hi = *(const v8bf*)(p + 8);
    qf[c] = __builtin_shufflevector(lo, hi, 0, 1, 2, 3, 4, 5, 6, 7,
                                    8, 9, 10, 11, 12, 13, 14, 15);
  }

  const v8f vzero = {0.f, 0.f, 0.f, 0.f, 0.f, 0.f, 0.f, 0.f};
  float rmax[8], rsum[8];
  v8f oacc[8];
#pragma unroll
  for (int r = 0; r < 8; ++r) { rmax[r] = -1e30f; rsum[r] = 0.f; }
#pragma unroll
  for (int nt = 0; nt < 8; ++nt) oacc[nt] = vzero;

  const float scale = 0.08838834764831845f;  // 1/sqrt(128)
  const int kend = qs + 16;                  // causal upper bound

  for (int kb0 = 0; kb0 < kend; kb0 += 32) {
    // ---- S = Q K^T for 32 keys (two 16x16 C tiles) ----
    v8f sc[2] = {vzero, vzero};
    const __bf16* kbase = Kb + ((long)(b * S_ + kb0) * NKV_ + hk) * HD_;
#pragma unroll
    for (int t = 0; t < 2; ++t)
#pragma unroll
      for (int c = 0; c < 4; ++c) {
        v16bf kf = *(const v16bf*)(kbase + (long)(t * 16 + lrow) * (NKV_ * HD_) +
                                   c * 32 + hi16 * 16);
        sc[t] = wmma_bf16(qf[c], kf, sc[t]);
      }

    // ---- scale + causal mask (C layout: row = r + hi16*8, col = lane&15) ----
#pragma unroll
    for (int t = 0; t < 2; ++t) {
      const int kidx = kb0 + t * 16 + lrow;
#pragma unroll
      for (int r = 0; r < 8; ++r) {
        const int qidx = qs + r + hi16 * 8;
        float v = sc[t][r] * scale;
        if (kidx > qidx) v += -1.0e9f;
        sc[t][r] = v;
      }
    }

    // ---- online softmax: per-row max/sum via 16-lane xor reductions ----
    float nmax[8], alpha[8];
#pragma unroll
    for (int r = 0; r < 8; ++r) {
      float m = fmaxf(sc[0][r], sc[1][r]);
      m = fmaxf(m, __shfl_xor(m, 1));
      m = fmaxf(m, __shfl_xor(m, 2));
      m = fmaxf(m, __shfl_xor(m, 4));
      m = fmaxf(m, __shfl_xor(m, 8));
      nmax[r]  = fmaxf(rmax[r], m);
      alpha[r] = __expf(rmax[r] - nmax[r]);
      rmax[r]  = nmax[r];
    }
#pragma unroll
    for (int r = 0; r < 8; ++r) {
      const float p0 = __expf(sc[0][r] - nmax[r]);
      const float p1 = __expf(sc[1][r] - nmax[r]);
      Ps[wid][r + hi16 * 8][lrow]      = f2bf(p0);
      Ps[wid][r + hi16 * 8][16 + lrow] = f2bf(p1);
      float ps = p0 + p1;
      ps += __shfl_xor(ps, 1);
      ps += __shfl_xor(ps, 2);
      ps += __shfl_xor(ps, 4);
      ps += __shfl_xor(ps, 8);
      rsum[r] = rsum[r] * alpha[r] + ps;
#pragma unroll
      for (int nt = 0; nt < 8; ++nt) oacc[nt][r] *= alpha[r];
    }

    // ---- re-fragment P (16x32) as WMMA A operand from LDS ----
    v8bf plo = *(const v8bf*)&Ps[wid][lrow][hi16 * 16];
    v8bf phi = *(const v8bf*)&Ps[wid][lrow][hi16 * 16 + 8];
    v16bf pf = __builtin_shufflevector(plo, phi, 0, 1, 2, 3, 4, 5, 6, 7,
                                       8, 9, 10, 11, 12, 13, 14, 15);

    // ---- O += P V : 8 n-tiles over head_dim, K=32 keys ----
    const __bf16* vbase =
        Vt + ((long)(b * NKV_ + hk) * HD_) * (long)S_ + kb0 + hi16 * 16;
#pragma unroll
    for (int nt = 0; nt < 8; ++nt) {
      v16bf vf = *(const v16bf*)(vbase + (long)(nt * 16 + lrow) * S_);
      oacc[nt] = wmma_bf16(pf, vf, oacc[nt]);
    }
  }

  // ---- normalize + store bf16, layout (b, q, h*128 + d) ----
  const long orow0 = (long)b * S_ + qs;
#pragma unroll
  for (int nt = 0; nt < 8; ++nt) {
    const long col = (long)h * HD_ + nt * 16 + lrow;
#pragma unroll
    for (int r = 0; r < 8; ++r) {
      const long row = orow0 + r + hi16 * 8;
      Ob[row * (NH_ * HD_) + col] = f2bf(oacc[nt][r] / rsum[r]);
    }
  }
}

// ---------------------------------------------------------------- launcher
extern "C" void kernel_launch(void* const* d_in, const int* in_sizes, int n_in,
                              void* d_out, int out_size, void* d_ws,
                              size_t ws_size, hipStream_t stream) {
  (void)in_sizes; (void)n_in; (void)out_size; (void)ws_size;
  const float* x    = (const float*)d_in[0];
  const float* wq   = (const float*)d_in[1];
  const float* wk   = (const float*)d_in[2];
  const float* wv   = (const float*)d_in[3];
  const float* wo   = (const float*)d_in[4];
  const float* fcos = (const float*)d_in[5];
  const float* fsin = (const float*)d_in[6];
  // d_in[7] = mask (recomputed in-kernel), d_in[8] = start_pos (0)

  const long MB  = (long)B_ * S_;        // 4096 rows
  const long KVN = (long)NKV_ * HD_;     // 1024

  char* ws = (char*)d_ws;
  size_t off = 0;
  auto alloc = [&](size_t bytes) -> void* {
    void* p = ws + off;
    off += (bytes + 255) & ~(size_t)255;
    return p;
  };
  __bf16* xb   = (__bf16*)alloc(MB * DIM_ * 2);
  __bf16* wqb  = (__bf16*)alloc((long)DIM_ * DIM_ * 2);
  __bf16* wkb  = (__bf16*)alloc(KVN * DIM_ * 2);
  __bf16* wvb  = (__bf16*)alloc(KVN * DIM_ * 2);
  __bf16* wob  = (__bf16*)alloc((long)DIM_ * DIM_ * 2);
  float*  qf32 = (float*)alloc(MB * DIM_ * 4);
  float*  kf32 = (float*)alloc(MB * KVN * 4);
  float*  vf32 = (float*)alloc(MB * KVN * 4);
  __bf16* qb   = (__bf16*)alloc(MB * DIM_ * 2);
  __bf16* kbb  = (__bf16*)alloc(MB * KVN * 2);
  __bf16* vtb  = (__bf16*)alloc(MB * KVN * 2);
  __bf16* ob   = (__bf16*)alloc(MB * DIM_ * 2);

  auto cvt = [&](const float* s, __bf16* d, long n) {
    int blocks = (int)((n + 255) / 256);
    if (blocks > 32768) blocks = 32768;
    convert_bf16<<<blocks, 256, 0, stream>>>(s, d, n);
  };
  cvt(x,  xb,  MB * DIM_);
  cvt(wq, wqb, (long)DIM_ * DIM_);
  cvt(wk, wkb, KVN * DIM_);
  cvt(wv, wvb, KVN * DIM_);
  cvt(wo, wob, (long)DIM_ * DIM_);

  // QKV projections (bf16 WMMA, f32 out)
  dim3 gq(DIM_ / 128, (unsigned)(MB / 128));
  gemm_bf16_nt<<<gq, 256, 0, stream>>>(xb, wqb, qf32, (int)MB, DIM_, DIM_);
  dim3 gkv((unsigned)(KVN / 128), (unsigned)(MB / 128));
  gemm_bf16_nt<<<gkv, 256, 0, stream>>>(xb, wkb, kf32, (int)MB, (int)KVN, DIM_);
  gemm_bf16_nt<<<gkv, 256, 0, stream>>>(xb, wvb, vf32, (int)MB, (int)KVN, DIM_);

  // RoPE + bf16 cast; V transpose + cast
  const long qpairs = MB * NH_ * 64;
  rope_to_bf16<<<(unsigned)((qpairs + 255) / 256), 256, 0, stream>>>(
      qf32, fcos, fsin, qb, NH_, qpairs);
  const long kpairs = MB * NKV_ * 64;
  rope_to_bf16<<<(unsigned)((kpairs + 255) / 256), 256, 0, stream>>>(
      kf32, fcos, fsin, kbb, NKV_, kpairs);
  const long vel = MB * KVN;
  v_transpose_bf16<<<(unsigned)((vel + 255) / 256), 256, 0, stream>>>(vf32, vtb,
                                                                      vel);

  // Flash attention: 8 q-tiles per block (one per wave)
  dim3 gfa(S_ / (16 * 8), NH_, B_);
  flash_attn_wmma<<<gfa, 256, 0, stream>>>(qb, kbb, vtb, ob);

  // Output projection straight into d_out (f32)
  dim3 go(DIM_ / 128, (unsigned)(MB / 128));
  gemm_bf16_nt<<<go, 256, 0, stream>>>(ob, wob, (float*)d_out, (int)MB, DIM_,
                                       DIM_);
}
